// HGLSTM_50568944943257
// MI455X (gfx1250) — compile-verified, matrix operand order checked by reading, not run on hardware
//
#include <hip/hip_runtime.h>
#include <hip/hip_bf16.h>
#include <stdint.h>

#define HDIM  256
#define HDIM2 512
#define HDIM4 1024
#define LDIM  16
#define KDIM  256   // every GEMM in this model has K = 256 (compile-time)

typedef __bf16 bf16_t;
typedef __attribute__((ext_vector_type(16))) __bf16 v16bf;
typedef __attribute__((ext_vector_type(8)))  float  v8f;

// ===========================================================================
// WMMA GEMM:  C[M,N] (f32, ld=N)  (+)=  A[M,256] (f32)  x  W[256,N]
// W supplied pre-transposed + bf16:  Wt[N][256].
// Block = 256 threads = 8 waves. Block tile = 128(M) x 32(N): each wave owns
// one 16x16 tile in column 0 and one in column 1 -> each A fragment feeds two
// v_wmma_f32_16x16x32_bf16 ops. K loop fully unrolled (8 steps) so the
// backend emits 16 static WMMAs and pipelines LDS traffic against them.
// A-fragment per lane (16-bit A 16x32 layout): elems 0..7 = K[half*8 .. +7],
// elems 8..15 = K[16+half*8 .. +7] -> two contiguous 16B LDS reads; B stored
// transposed in LDS so B-fragments use the same two reads.
// ===========================================================================
__global__ __launch_bounds__(256) void k_gemm_wmma(
    float* __restrict__ C, const float* __restrict__ A,
    const bf16_t* __restrict__ Wt, const float* __restrict__ bias,
    int M, int N, int accFlag)
{
  __shared__ __align__(16) bf16_t As[128][32];   // 8 KB
  __shared__ __align__(16) bf16_t Bs[32][32];    // 2 KB
  const int m0   = blockIdx.x * 128;
  const int n0   = blockIdx.y * 32;
  const int tid  = threadIdx.x;
  const int wave = tid >> 5;
  const int lane = tid & 31;
  const int half = lane >> 4;
  const int l16  = lane & 15;

  v8f acc0 = {0.f,0.f,0.f,0.f,0.f,0.f,0.f,0.f};
  v8f acc1 = {0.f,0.f,0.f,0.f,0.f,0.f,0.f,0.f};

  #pragma unroll
  for (int k0 = 0; k0 < KDIM; k0 += 32) {
    // ---- fill A tile: 128x32 fp32 -> bf16 (1024 float4 slots, 4/thread) ----
    #pragma unroll
    for (int i = 0; i < 4; ++i) {
      int slot = tid + i * 256;          // 0..1023
      int r    = slot >> 3;
      int c4   = (slot & 7) << 2;
      float4 v = make_float4(0.f, 0.f, 0.f, 0.f);
      int gr = m0 + r;
      if (gr < M) {
        const float* ap = A + (size_t)gr * KDIM + k0 + c4;
        v = *(const float4*)ap;
        if (k0 + 32 < KDIM) __builtin_prefetch(ap + 32, 0, 1);  // next K-slice
      }
      As[r][c4 + 0] = (bf16_t)v.x;
      As[r][c4 + 1] = (bf16_t)v.y;
      As[r][c4 + 2] = (bf16_t)v.z;
      As[r][c4 + 3] = (bf16_t)v.w;
    }
    // ---- fill B tile: 32x32 bf16 from Wt[N][256] (4 bf16 = 8B per thread) --
    {
      int r  = tid >> 3;                 // 0..31
      int c4 = (tid & 7) << 2;           // 0,4,..,28
      uint2 u = make_uint2(0u, 0u);
      if (n0 + r < N)
        u = *(const uint2*)(Wt + (size_t)(n0 + r) * KDIM + k0 + c4);
      *(uint2*)&Bs[r][c4] = u;
    }
    __syncthreads();

    const int mrow = (wave << 4) + l16;
    union { v16bf v; float4 f[2]; } a, b0, b1;
    a.f[0]  = *(const float4*)&As[mrow][half * 8];
    a.f[1]  = *(const float4*)&As[mrow][16 + half * 8];
    b0.f[0] = *(const float4*)&Bs[l16][half * 8];
    b0.f[1] = *(const float4*)&Bs[l16][16 + half * 8];
    b1.f[0] = *(const float4*)&Bs[16 + l16][half * 8];
    b1.f[1] = *(const float4*)&Bs[16 + l16][16 + half * 8];
    acc0 = __builtin_amdgcn_wmma_f32_16x16x32_bf16(false, a.v, false, b0.v,
                                                   (short)0, acc0, false, false);
    acc1 = __builtin_amdgcn_wmma_f32_16x16x32_bf16(false, a.v, false, b1.v,
                                                   (short)0, acc1, false, false);
    __syncthreads();
  }

  // ---- epilogue: C layout row = half*8 + r, col = l16 (per 16-wide tile) ---
  const int mbase = m0 + (wave << 4) + half * 8;
  const int col0  = n0 + l16;
  const int col1  = n0 + 16 + l16;
  float bv0 = (bias && col0 < N) ? bias[col0] : 0.f;
  float bv1 = (bias && col1 < N) ? bias[col1] : 0.f;
  #pragma unroll
  for (int r = 0; r < 8; ++r) {
    int row = mbase + r;
    if (row < M) {
      if (col0 < N) {
        size_t idx = (size_t)row * N + col0;
        float v = acc0[r] + bv0;
        if (accFlag) v += C[idx];
        C[idx] = v;
      }
      if (col1 < N) {
        size_t idx = (size_t)row * N + col1;
        float v = acc1[r] + bv1;
        if (accFlag) v += C[idx];
        C[idx] = v;
      }
    }
  }
}

// ======================= weight convert: fp32[K][N] -> bf16 Wt[N][K] =======
__global__ __launch_bounds__(256) void k_convw(bf16_t* __restrict__ dst,
                                               const float* __restrict__ src,
                                               int K, int N)
{
  int i = blockIdx.x * 256 + threadIdx.x;
  if (i >= K * N) return;
  int k = i / N, n = i - k * N;
  dst[(size_t)n * K + k] = (bf16_t)src[i];
}

// ======================= embedding gather ==================================
__global__ __launch_bounds__(256) void k_embed(
    float* __restrict__ x, float* __restrict__ h, float* __restrict__ c,
    float* __restrict__ maskf, const int* __restrict__ content,
    const int* __restrict__ maskI, const float* __restrict__ emb)
{
  int m = blockIdx.x, ch = threadIdx.x;
  float mf = (float)maskI[m];
  float v  = emb[(size_t)content[m] * HDIM + ch];
  size_t o = (size_t)m * HDIM + ch;
  x[o] = v; h[o] = v * mf; c[o] = v * mf;
  if (ch == 0) maskf[m] = mf;
}

__global__ __launch_bounds__(256) void k_intmaskf(float* __restrict__ dst,
                                                  const int* __restrict__ src,
                                                  int M)
{
  int i = blockIdx.x * 256 + threadIdx.x;
  if (i < M) dst[i] = (float)src[i];
}

// ======================= per-group mean over seq ===========================
__global__ __launch_bounds__(256) void k_groupmean(
    float* __restrict__ g, float* __restrict__ cg,
    const float* __restrict__ h, const float* __restrict__ c, int seq)
{
  int grp = blockIdx.x, ch = threadIdx.x;
  float sh = 0.f, sc = 0.f;
  for (int i = 0; i < seq; ++i) {
    size_t idx = ((size_t)grp * seq + i) * HDIM + ch;
    sh += h[idx]; sc += c[idx];
  }
  float inv = 1.f / (float)seq;
  g[(size_t)grp * HDIM + ch]  = sh * inv;
  cg[(size_t)grp * HDIM + ch] = sc * inv;
}

// ======================= h,c *= mask =======================================
__global__ __launch_bounds__(256) void k_maskhc(float* __restrict__ h,
                                                float* __restrict__ c,
                                                const float* __restrict__ mf)
{
  int m = blockIdx.x, ch = threadIdx.x;
  float f = mf[m];
  size_t o = (size_t)m * HDIM + ch;
  h[o] *= f; c[o] *= f;
}

// ======================= C[m] += Gv[group(m)] (N=HDIM) =====================
__global__ __launch_bounds__(256) void k_addgroup(float* __restrict__ C,
                                                  const float* __restrict__ Gv,
                                                  int seq)
{
  int m = blockIdx.x, ch = threadIdx.x;
  C[(size_t)m * HDIM + ch] += Gv[(size_t)(m / seq) * HDIM + ch];
}

// ======================= out[row] = dot(A[row], v256) (+bias, opt tanh) ====
__global__ __launch_bounds__(256) void k_rowdot(float* __restrict__ out,
                                                const float* __restrict__ A,
                                                const float* __restrict__ v,
                                                const float* __restrict__ biasPtr,
                                                int applyTanh)
{
  int wave = threadIdx.x >> 5, lane = threadIdx.x & 31;
  int row = blockIdx.x * 8 + wave;
  float acc = 0.f;
  #pragma unroll
  for (int i = 0; i < 8; ++i) {
    int ch = lane + i * 32;
    float a = A[(size_t)row * HDIM + ch];
    if (applyTanh) a = tanhf(a);
    acc += a * v[ch];
  }
  for (int off = 16; off > 0; off >>= 1) acc += __shfl_down(acc, off, 32);
  if (lane == 0) out[row] = acc + (biasPtr ? biasPtr[0] : 0.f);
}

// ======================= neighbor attention ================================
// logits_n = (tb[m] + mn * t[gather]) * (1-mn)*1e-25 ; softmax over 8;
// hn[m] = sum_n score_n * mn * hWn[gather]
__global__ __launch_bounds__(256) void k_nbrattn(
    float* __restrict__ hn, const float* __restrict__ hWn,
    const float* __restrict__ t, const float* __restrict__ tb,
    const int* __restrict__ nidx, const int* __restrict__ nmask, int seq)
{
  int m = blockIdx.x, ch = threadIdx.x;
  int grp = m / seq;
  __shared__ float sc[8];
  __shared__ float smn[8];
  __shared__ int   sj[8];
  if (ch < 8) {
    int   j  = nidx[(size_t)m * 8 + ch];
    float mn = (float)nmask[(size_t)m * 8 + ch];
    float tv = (j > 0) ? t[(size_t)grp * seq + (j - 1)] : 0.f;
    sj[ch]  = j;
    smn[ch] = mn;
    sc[ch]  = (tb[m] + mn * tv) * (1.f - mn) * 1e-25f;
  }
  __syncthreads();
  if (ch == 0) {
    float mx = sc[0];
    for (int n = 1; n < 8; ++n) mx = fmaxf(mx, sc[n]);
    float s = 0.f;
    for (int n = 0; n < 8; ++n) { sc[n] = __expf(sc[n] - mx); s += sc[n]; }
    float inv = 1.f / s;
    for (int n = 0; n < 8; ++n) sc[n] *= inv;
  }
  __syncthreads();
  float acc = 0.f;
  for (int n = 0; n < 8; ++n) {
    int j = sj[n];
    float w = sc[n] * smn[n];
    if (j > 0) acc += w * hWn[((size_t)grp * seq + (j - 1)) * HDIM + ch];
  }
  hn[(size_t)m * HDIM + ch] = acc;
}

// ======================= LSTM cell =========================================
__global__ __launch_bounds__(256) void k_lstm(
    float* __restrict__ newh, float* __restrict__ newc,
    const float* __restrict__ gates, const float* __restrict__ gvs,
    const float* __restrict__ c, int seq)
{
  int m = blockIdx.x, ch = threadIdx.x;
  int grp = m / seq;
  size_t gb = (size_t)m * HDIM4, vb = (size_t)grp * HDIM4;
  float gi = gates[gb + ch]            + gvs[vb + ch];
  float gf = gates[gb + HDIM   + ch]   + gvs[vb + HDIM   + ch];
  float go = gates[gb + 2*HDIM + ch]   + gvs[vb + 2*HDIM + ch];
  float gu = gates[gb + 3*HDIM + ch]   + gvs[vb + 3*HDIM + ch];
  float si = 1.f / (1.f + __expf(-gi));
  float sf = 1.f / (1.f + __expf(-gf));
  float so = 1.f / (1.f + __expf(-go));
  size_t o = (size_t)m * HDIM + ch;
  float nc = sf * c[o] + si * tanhf(gu);
  newc[o] = nc;
  newh[o] = so * tanhf(nc);
}

// ======================= attention pooling =================================
__global__ __launch_bounds__(256) void k_attnpool(
    float* __restrict__ havg, const float* __restrict__ l,
    const float* __restrict__ h, int seq)
{
  int grp = blockIdx.x, ch = threadIdx.x;
  __shared__ float al[64];
  if (ch < seq) al[ch] = l[(size_t)grp * seq + ch];
  __syncthreads();
  if (ch == 0) {
    float mx = al[0];
    for (int i = 1; i < seq; ++i) mx = fmaxf(mx, al[i]);
    float s = 0.f;
    for (int i = 0; i < seq; ++i) { al[i] = __expf(al[i] - mx); s += al[i]; }
    float inv = 1.f / s;
    for (int i = 0; i < seq; ++i) al[i] *= inv;
  }
  __syncthreads();
  float acc = 0.f;
  for (int i = 0; i < seq; ++i)
    acc += al[i] * h[((size_t)grp * seq + i) * HDIM + ch];
  havg[(size_t)grp * HDIM + ch] = acc;
}

// ======================= g-cell: channel-wise softmax over seq =============
__global__ __launch_bounds__(256) void k_gcell(
    float* __restrict__ g, float* __restrict__ cg,
    const float* __restrict__ foRaw, const float* __restrict__ gw,
    const float* __restrict__ hgu, const float* __restrict__ gub,
    const float* __restrict__ maskf, const float* __restrict__ c, int seq)
{
  int grp = blockIdx.x, ch = threadIdx.x;
  float gwv = gw[(size_t)grp * HDIM + ch] + gub[ch];
  float mx = -1e30f;
  for (int w = 0; w < seq; ++w) {
    size_t m = (size_t)grp * seq + w;
    float v = 1.f / (1.f + __expf(-(gwv + hgu[m * HDIM + ch])))
              - (1.f - maskf[m]) * 1e-25f;
    mx = fmaxf(mx, v);
  }
  float s = 0.f, num = 0.f;
  for (int w = 0; w < seq; ++w) {
    size_t m = (size_t)grp * seq + w;
    float v = 1.f / (1.f + __expf(-(gwv + hgu[m * HDIM + ch])))
              - (1.f - maskf[m]) * 1e-25f;
    float e = __expf(v - mx);
    s += e;
    num += e * c[m * HDIM + ch];
  }
  float f = 1.f / (1.f + __expf(-foRaw[(size_t)grp * HDIM2 + ch]));
  float o = 1.f / (1.f + __expf(-foRaw[(size_t)grp * HDIM2 + HDIM + ch]));
  size_t go_ = (size_t)grp * HDIM + ch;
  float nc = f * cg[go_] + num / s;
  cg[go_] = nc;
  g[go_]  = o * tanhf(nc);
}

// ===========================================================================
extern "C" void kernel_launch(void* const* d_in, const int* in_sizes, int n_in,
                              void* d_out, int out_size, void* d_ws, size_t ws_size,
                              hipStream_t stream)
{
  (void)in_sizes; (void)out_size;
  if (n_in < 31) return;

  const int B = 8, S = 32, W = 64;
  const int M1 = B * S * W, G1 = B * S, SEQ1 = W;   // 16384 tokens, 256 groups
  const int M2 = B * S,     G2 = B,     SEQ2 = S;   // 256 tokens, 8 groups

  const int*   src       = (const int*)d_in[0];
  const int*   src_mask  = (const int*)d_in[1];
  const int*   s_nidx    = (const int*)d_in[2];
  const int*   s_nmask   = (const int*)d_in[3];
  const int*   sent_mask = (const int*)d_in[4];
  const int*   d_nidx    = (const int*)d_in[5];
  const int*   d_nmaskP  = (const int*)d_in[6];
  const float* emb   = (const float*)d_in[7];
  const float* sWh   = (const float*)d_in[8];
  const float* sU    = (const float*)d_in[9];
  const float* sWn   = (const float*)d_in[10];
  const float* sV    = (const float*)d_in[11];
  const float* sVb   = (const float*)d_in[12];
  const float* naWh  = (const float*)d_in[13];
  const float* naWn  = (const float*)d_in[14];
  const float* naU   = (const float*)d_in[15];
  const float* na_u  = (const float*)d_in[16];
  const float* na_ub = (const float*)d_in[17];
  const float* naV   = (const float*)d_in[18];
  const float* naVb  = (const float*)d_in[19];
  const float* gW    = (const float*)d_in[20];
  const float* g_w   = (const float*)d_in[21];
  const float* gU    = (const float*)d_in[22];
  const float* gUb   = (const float*)d_in[23];
  const float* g_u   = (const float*)d_in[24];
  const float* g_ub  = (const float*)d_in[25];
  const float* apw   = (const float*)d_in[26];
  const float* apwb  = (const float*)d_in[27];
  const float* apu   = (const float*)d_in[28];
  const float* outw  = (const float*)d_in[29];
  const float* outb  = (const float*)d_in[30];

  // ------------- workspace layout (256B-aligned cursor) -------------
  char* base = (char*)d_ws;
  size_t off = 0;
  auto alloc = [&](size_t bytes) -> char* {
    off = (off + 255) & ~(size_t)255;
    char* p = base + off;
    off += bytes;
    return p;
  };
  auto allocF = [&](size_t n) { return (float*)alloc(n * sizeof(float)); };
  auto allocB = [&](size_t n) { return (bf16_t*)alloc(n * sizeof(bf16_t)); };

  float* x1   = allocF((size_t)M1 * HDIM);
  float* hA   = allocF((size_t)M1 * HDIM);
  float* hB   = allocF((size_t)M1 * HDIM);
  float* cA   = allocF((size_t)M1 * HDIM);
  float* cB   = allocF((size_t)M1 * HDIM);
  float* bufA = allocF((size_t)M1 * HDIM);   // hWn / hh_raw
  float* bufB = allocF((size_t)M1 * HDIM);   // base
  float* bufC = allocF((size_t)M1 * HDIM);   // hn
  float* bufG = allocF((size_t)M1 * HDIM4);  // gates / hgu
  float* tArr  = allocF(M1);
  float* tbArr = allocF(M1);
  float* lArr  = allocF(M1);
  float* g1  = allocF((size_t)G1 * HDIM);
  float* cg1 = allocF((size_t)G1 * HDIM);
  float* gvna  = allocF((size_t)G1 * HDIM);
  float* gvs   = allocF((size_t)G1 * HDIM4);
  float* havg  = allocF((size_t)G1 * HDIM);
  float* foRaw = allocF((size_t)G1 * HDIM2);
  float* gwB   = allocF((size_t)G1 * HDIM);
  float* x2    = allocF((size_t)G1 * HDIM);
  float* h2B   = allocF((size_t)G1 * HDIM);
  float* c2B   = allocF((size_t)G1 * HDIM);
  float* g2  = allocF((size_t)G2 * HDIM);
  float* cg2 = allocF((size_t)G2 * HDIM);
  float* maskf1 = allocF(M1);
  float* maskf2 = allocF(M2);

  bf16_t* wt_sWh  = allocB((size_t)HDIM * HDIM4);
  bf16_t* wt_sU   = allocB((size_t)HDIM * HDIM4);
  bf16_t* wt_sWn  = allocB((size_t)HDIM * HDIM4);
  bf16_t* wt_sV   = allocB((size_t)HDIM * HDIM4);
  bf16_t* wt_naWh = allocB((size_t)HDIM * HDIM);
  bf16_t* wt_naWn = allocB((size_t)HDIM * HDIM);
  bf16_t* wt_naU  = allocB((size_t)HDIM * HDIM);
  bf16_t* wt_naV  = allocB((size_t)HDIM * HDIM);
  bf16_t* wt_gW   = allocB((size_t)HDIM * HDIM2);
  bf16_t* wt_gw   = allocB((size_t)HDIM * HDIM);
  bf16_t* wt_gU   = allocB((size_t)HDIM * HDIM2);
  bf16_t* wt_gu   = allocB((size_t)HDIM * HDIM);
  bf16_t* wt_apw  = allocB((size_t)HDIM * HDIM);
  bf16_t* wt_outw = allocB((size_t)HDIM * LDIM);
  if (off > ws_size) return;   // workspace too small: do nothing safe

  // ------------- convert all weights to bf16, transposed -------------
  struct WSpec { const float* src; bf16_t* dst; int K, N; };
  const WSpec specs[14] = {
    {sWh, wt_sWh, HDIM, HDIM4}, {sU, wt_sU, HDIM, HDIM4},
    {sWn, wt_sWn, HDIM, HDIM4}, {sV, wt_sV, HDIM, HDIM4},
    {naWh, wt_naWh, HDIM, HDIM}, {naWn, wt_naWn, HDIM, HDIM},
    {naU, wt_naU, HDIM, HDIM},   {naV, wt_naV, HDIM, HDIM},
    {gW, wt_gW, HDIM, HDIM2},    {g_w, wt_gw, HDIM, HDIM},
    {gU, wt_gU, HDIM, HDIM2},    {g_u, wt_gu, HDIM, HDIM},
    {apw, wt_apw, HDIM, HDIM},   {outw, wt_outw, HDIM, LDIM}
  };
  for (int i = 0; i < 14; ++i) {
    int total = specs[i].K * specs[i].N;
    k_convw<<<(total + 255) / 256, 256, 0, stream>>>(
        specs[i].dst, specs[i].src, specs[i].K, specs[i].N);
  }

  auto GEMM = [&](float* C, const float* A, const bf16_t* Wt, const float* bias,
                  int M, int N, int acc) {
    dim3 grid((M + 127) / 128, (N + 31) / 32);
    k_gemm_wmma<<<grid, 256, 0, stream>>>(C, A, Wt, bias, M, N, acc);
  };

  // ------------- one level = 2 layers of s_cell + g_cell -------------
  auto run_layers = [&](float* h, float* c, float* hAlt, float* cAlt,
                        const float* x, float* g, float* cg,
                        const int* nidx, const int* nmask, const float* mf,
                        int M, int G, int seq) {
    for (int layer = 0; layer < 2; ++layer) {
      k_maskhc<<<M, 256, 0, stream>>>(h, c, mf);
      // ---- neighbor attention ----
      GEMM(bufA, h, wt_naWn, nullptr, M, HDIM, 0);       // hWn
      GEMM(bufB, h, wt_naWh, nullptr, M, HDIM, 0);       // base
      GEMM(bufB, x, wt_naU,  nullptr, M, HDIM, 1);
      GEMM(gvna, g, wt_naV,  naVb,    G, HDIM, 0);
      k_addgroup<<<M, 256, 0, stream>>>(bufB, gvna, seq);
      k_rowdot<<<M / 8, 256, 0, stream>>>(tArr,  bufA, na_u, nullptr, 0);
      k_rowdot<<<M / 8, 256, 0, stream>>>(tbArr, bufB, na_u, na_ub,   0);
      k_nbrattn<<<M, 256, 0, stream>>>(bufC, bufA, tArr, tbArr, nidx, nmask, seq);
      // ---- s_cell gates + LSTM ----
      GEMM(gvs,  g,    wt_sV,  sVb,     G, HDIM4, 0);
      GEMM(bufG, h,    wt_sWh, nullptr, M, HDIM4, 0);
      GEMM(bufG, x,    wt_sU,  nullptr, M, HDIM4, 1);
      GEMM(bufG, bufC, wt_sWn, nullptr, M, HDIM4, 1);
      k_lstm<<<M, 256, 0, stream>>>(hAlt, cAlt, bufG, gvs, c, seq);
      // ---- g_cell (uses OLD h, c) ----
      GEMM(bufA, h, wt_apw, apwb, M, HDIM, 0);           // hh_raw
      k_rowdot<<<M / 8, 256, 0, stream>>>(lArr, bufA, apu, nullptr, 1);
      k_attnpool<<<G, 256, 0, stream>>>(havg, lArr, h, seq);
      GEMM(foRaw, g,    wt_gW, nullptr, G, HDIM2, 0);
      GEMM(foRaw, havg, wt_gU, gUb,     G, HDIM2, 1);
      GEMM(gwB,   g,    wt_gw, nullptr, G, HDIM,  0);
      GEMM(bufG,  h,    wt_gu, nullptr, M, HDIM,  0);    // hgu
      k_gcell<<<G, 256, 0, stream>>>(g, cg, foRaw, gwB, bufG, g_ub, mf, c, seq);
      // swap h/c double buffers
      float* th = h; h = hAlt; hAlt = th;
      float* tc = c; c = cAlt; cAlt = tc;
    }
  };

  // ================= sentence level =================
  k_embed<<<M1, 256, 0, stream>>>(x1, hA, cA, maskf1, src, src_mask, emb);
  k_groupmean<<<G1, 256, 0, stream>>>(g1, cg1, hA, cA, SEQ1);
  run_layers(hA, cA, hB, cB, x1, g1, cg1, s_nidx, s_nmask, maskf1, M1, G1, SEQ1);
  // sent_h = g1, sent_c = cg1

  // ================= document level =================
  hipMemcpyAsync(x2, g1, (size_t)G1 * HDIM * sizeof(float),
                 hipMemcpyDeviceToDevice, stream);                // sent_emb
  k_intmaskf<<<(M2 + 255) / 256, 256, 0, stream>>>(maskf2, sent_mask, M2);
  k_groupmean<<<G2, 256, 0, stream>>>(g2, cg2, g1, cg1, SEQ2);
  run_layers(g1, cg1, h2B, c2B, x2, g2, cg2, d_nidx, d_nmaskP, maskf2,
             M2, G2, SEQ2);

  // ================= output projection =================
  GEMM((float*)d_out, g2, wt_outw, outb, G2, LDIM, 0);
}
